// BottleneckAdapter_67276367724646
// MI455X (gfx1250) — compile-verified, matrix operand order checked by reading, not run on hardware
//
#include <hip/hip_runtime.h>
#include <hip/hip_bf16.h>

// ---------------------------------------------------------------------------
// BottleneckAdapter on gfx1250 (MI455X): bf16 WMMA + TDM (tensor_load_to_lds)
//   B=16, S=2048 -> M = 32768 rows; C=768, Q=1024, D=64.
// K1 streams the f32 activations directly via TDM (data_size=4B) and converts
// to bf16 in-register, avoiding a whole-tensor f32->bf16 pre-pass.
// ---------------------------------------------------------------------------

typedef __attribute__((ext_vector_type(16))) __bf16 bf16x16;
typedef __attribute__((ext_vector_type(8)))  float  f32x8;
typedef __attribute__((ext_vector_type(4)))  unsigned int u32x4;
typedef __attribute__((ext_vector_type(8)))  unsigned int u32x8;

#define M_TOT 32768
#define C_DIM 768
#define Q_DIM 1024
#define D_DIM 64
#define EPS   1e-5f

union Frag {
    bf16x16 v;
    uint4   q[2];
};

// 16x32 bf16 operand fragment, ISA 7.12.2 16-bit A layout:
//   lane L (L<16):  row = base + L,    halves 0..7 -> K += 0..7,  halves 8..15 -> K += 16..23
//   lane L (L>=16): row = base + L-16, halves 0..7 -> K += 8..15, halves 8..15 -> K += 24..31
__device__ __forceinline__ Frag load_frag(const __hip_bfloat16* p0, int ld, int row, int k0) {
    const int lane = threadIdx.x & 31;
    const int r    = row + (lane & 15);
    const int ks   = (lane >> 4) << 3;            // 0 or 8
    const __hip_bfloat16* p = p0 + (size_t)r * ld + (k0 + ks);
    Frag f;
    f.q[0] = *(const uint4*)(p);
    f.q[1] = *(const uint4*)(p + 16);
    return f;
}

// Same fragment layout, but source is f32 in LDS: convert to bf16 on the fly.
__device__ __forceinline__ Frag load_frag_f32(const float* p0, int ld, int row) {
    const int lane = threadIdx.x & 31;
    const int r    = row + (lane & 15);
    const int ks   = (lane >> 4) << 3;            // 0 or 8
    const float* p = p0 + r * ld + ks;
    float4 x0 = *(const float4*)(p);
    float4 x1 = *(const float4*)(p + 4);
    float4 y0 = *(const float4*)(p + 16);
    float4 y1 = *(const float4*)(p + 20);
    Frag f;
    f.v[0]  = (__bf16)x0.x; f.v[1]  = (__bf16)x0.y; f.v[2]  = (__bf16)x0.z; f.v[3]  = (__bf16)x0.w;
    f.v[4]  = (__bf16)x1.x; f.v[5]  = (__bf16)x1.y; f.v[6]  = (__bf16)x1.z; f.v[7]  = (__bf16)x1.w;
    f.v[8]  = (__bf16)y0.x; f.v[9]  = (__bf16)y0.y; f.v[10] = (__bf16)y0.z; f.v[11] = (__bf16)y0.w;
    f.v[12] = (__bf16)y1.x; f.v[13] = (__bf16)y1.y; f.v[14] = (__bf16)y1.z; f.v[15] = (__bf16)y1.w;
    return f;
}

__device__ __forceinline__ f32x8 wmma_bf16(const Frag& a, const Frag& b, f32x8 c) {
    return __builtin_amdgcn_wmma_f32_16x16x32_bf16(
        false, a.v, false, b.v, (short)0, c, false, false);
}

// ---------------------------------------------------------------------------
// Tensor Data Mover: 2D tile load Global -> LDS (D# per ISA 08).
//   ds_code: 0=1B 1=2B 2=4B 3=8B elements; dims in elements.
// ---------------------------------------------------------------------------
__device__ __forceinline__ void tdm_load_2d(unsigned lds_byte_addr,
                                            const void* gptr,
                                            unsigned tile_cols, unsigned tile_rows,
                                            unsigned row_len,   unsigned n_rows,
                                            unsigned stride_elems, unsigned ds_code) {
    const unsigned long long ga = (unsigned long long)(uintptr_t)gptr;
    u32x4 g0;
    g0[0] = 1u;                                                // count=1, user mode
    g0[1] = lds_byte_addr;                                     // lds_addr
    g0[2] = (unsigned)ga;                                      // global_addr[31:0]
    g0[3] = (unsigned)((ga >> 32) & 0x01ffffffu) | (2u << 30); // ga[56:32] | type=2
    u32x8 g1;
    g1[0] = ds_code << 16;                                     // data_size
    g1[1] = (row_len & 0xffffu) << 16;                         // tensor_dim0[15:0]
    g1[2] = (row_len >> 16) | ((n_rows & 0xffffu) << 16);      // dim0[31:16]|dim1[15:0]
    g1[3] = (n_rows >> 16) | (tile_cols << 16);                // dim1[31:16]|tile_dim0
    g1[4] = tile_rows;                                         // tile_dim1 (tile_dim2=0)
    g1[5] = stride_elems;                                      // tensor_dim0_stride[31:0]
    g1[6] = 0u;
    g1[7] = 0u;
    asm volatile("tensor_load_to_lds %0, %1" :: "s"(g0), "s"(g1) : "memory");
}

// ---------------------------------------------------------------------------
// f32 -> bf16 conversion (weights only)
// ---------------------------------------------------------------------------
__global__ void cvt_f32_bf16(const float* __restrict__ in, __hip_bfloat16* __restrict__ out, int n) {
    int i = blockIdx.x * blockDim.x + threadIdx.x;
    if (i < n) out[i] = __float2bfloat16(in[i]);
}

// ---------------------------------------------------------------------------
// K1: projected[M,Q] = A[M,C](f32) x Wproj[Q,C]^T(bf16)  -> bf16
// grid = (M/128, Q/128), block = 256.  TDM double-buffers a 128x32 f32 A-tile
// (16 KB) and a 128x32 bf16 B-tile (8 KB); A converts f32->bf16 at frag build.
// ---------------------------------------------------------------------------
__global__ void k1_proj(const float* __restrict__ A,
                        const __hip_bfloat16* __restrict__ W,
                        __hip_bfloat16* __restrict__ P) {
    __shared__ alignas(16) float          sAf[2 * 128 * 32];   // 32 KB
    __shared__ alignas(16) __hip_bfloat16 sB [2 * 128 * 32];   // 16 KB

    const int wave = threadIdx.x >> 5;
    const int lane = threadIdx.x & 31;
    const int m0   = blockIdx.x * 128 + wave * 16;
    const int n0b  = blockIdx.y * 128;

    const unsigned ldsA = (unsigned)(uintptr_t)sAf;
    const unsigned ldsB = (unsigned)(uintptr_t)sB;
    const int nsteps = C_DIM / 32;

    if (wave == 0) {
        tdm_load_2d(ldsA, A + (size_t)blockIdx.x * 128 * C_DIM, 32, 128, C_DIM, M_TOT, C_DIM, 2);
        tdm_load_2d(ldsB, W + (size_t)n0b * C_DIM,              32, 128, C_DIM, Q_DIM, C_DIM, 1);
    }

    f32x8 acc[8] = {};
    for (int ki = 0; ki < nsteps; ++ki) {
        const int buf = ki & 1;
        if (wave == 0) {
            if (ki + 1 < nsteps) {
                tdm_load_2d(ldsA + (unsigned)((buf ^ 1) * 128 * 32 * 4),
                            A + (size_t)blockIdx.x * 128 * C_DIM + (ki + 1) * 32,
                            32, 128, C_DIM, M_TOT, C_DIM, 2);
                tdm_load_2d(ldsB + (unsigned)((buf ^ 1) * 128 * 32 * 2),
                            W + (size_t)n0b * C_DIM + (ki + 1) * 32,
                            32, 128, C_DIM, Q_DIM, C_DIM, 1);
                __builtin_amdgcn_s_wait_tensorcnt(2);
            } else {
                __builtin_amdgcn_s_wait_tensorcnt(0);
            }
        }
        __syncthreads();
        Frag a = load_frag_f32(sAf + buf * 128 * 32, 32, wave * 16);
#pragma unroll
        for (int t = 0; t < 8; ++t) {
            Frag b = load_frag(sB + buf * 128 * 32, 32, 16 * t, 0);
            acc[t] = wmma_bf16(a, b, acc[t]);
        }
        __syncthreads();
    }

    const int mb = m0 + ((lane >> 4) << 3);
#pragma unroll
    for (int t = 0; t < 8; ++t) {
        const int n = n0b + 16 * t + (lane & 15);
#pragma unroll
        for (int r = 0; r < 8; ++r)
            P[(size_t)(mb + r) * Q_DIM + n] = __float2bfloat16(acc[t][r]);
    }
}

// ---------------------------------------------------------------------------
// bf16/bf16 TDM mainloop for K2: block tile 128(M) x 64(N), K steps of 32.
// ---------------------------------------------------------------------------
__device__ __forceinline__ void gemm128x64_tdm(const __hip_bfloat16* A, int lda, int m0blk,
                                               const __hip_bfloat16* B, int ldb,
                                               int kdim,
                                               __hip_bfloat16* sA, __hip_bfloat16* sB,
                                               f32x8 acc[4]) {
    const int wave   = threadIdx.x >> 5;
    const int nsteps = kdim / 32;
    const unsigned ldsA = (unsigned)(uintptr_t)sA;
    const unsigned ldsB = (unsigned)(uintptr_t)sB;

    if (wave == 0) {
        tdm_load_2d(ldsA, A + (size_t)m0blk * lda, 32, 128, lda, M_TOT, lda, 1);
        tdm_load_2d(ldsB, B,                       32, 64,  ldb, D_DIM, ldb, 1);
    }
    for (int ki = 0; ki < nsteps; ++ki) {
        const int buf = ki & 1;
        if (wave == 0) {
            if (ki + 1 < nsteps) {
                tdm_load_2d(ldsA + (unsigned)((buf ^ 1) * 128 * 32 * 2),
                            A + (size_t)m0blk * lda + (ki + 1) * 32,
                            32, 128, lda, M_TOT, lda, 1);
                tdm_load_2d(ldsB + (unsigned)((buf ^ 1) * 64 * 32 * 2),
                            B + (ki + 1) * 32,
                            32, 64, ldb, D_DIM, ldb, 1);
                __builtin_amdgcn_s_wait_tensorcnt(2);
            } else {
                __builtin_amdgcn_s_wait_tensorcnt(0);
            }
        }
        __syncthreads();
        Frag a = load_frag(sA + buf * 128 * 32, 32, wave * 16, 0);
#pragma unroll
        for (int t = 0; t < 4; ++t) {
            Frag b = load_frag(sB + buf * 64 * 32, 32, 16 * t, 0);
            acc[t] = wmma_bf16(a, b, acc[t]);
        }
        __syncthreads();
    }
}

// ---------------------------------------------------------------------------
// K2 (fused): down = P x Wdown^T (TDM) ; LayerNorm ; SwiGLU from LDS
// grid = (M/128), block = 256
// ---------------------------------------------------------------------------
__global__ void k2_down_ln_swiglu(const __hip_bfloat16* __restrict__ P,
                                  const __hip_bfloat16* __restrict__ Wd,
                                  const float* __restrict__ gamma,
                                  const float* __restrict__ beta,
                                  const __hip_bfloat16* __restrict__ W1,
                                  const float* __restrict__ b1,
                                  const __hip_bfloat16* __restrict__ W2,
                                  const float* __restrict__ b2,
                                  __hip_bfloat16* __restrict__ Act) {
    __shared__ alignas(16) __hip_bfloat16 sA[2 * 128 * 32];
    __shared__ alignas(16) __hip_bfloat16 sB[2 * 64 * 32];
    __shared__ float s_down[128][68];
    __shared__ alignas(16) __hip_bfloat16 s_norm[128 * 64];

    const int wave = threadIdx.x >> 5;
    const int lane = threadIdx.x & 31;
    const int m0   = blockIdx.x * 128 + wave * 16;

    f32x8 acc[4] = {};
    gemm128x64_tdm(P, Q_DIM, blockIdx.x * 128, Wd, Q_DIM, Q_DIM, sA, sB, acc);

    {
        const int ml = wave * 16 + ((lane >> 4) << 3);
#pragma unroll
        for (int t = 0; t < 4; ++t)
#pragma unroll
            for (int r = 0; r < 8; ++r)
                s_down[ml + r][16 * t + (lane & 15)] = acc[t][r];
    }
    __syncthreads();

    if (threadIdx.x < 128) {
        const int row = threadIdx.x;
        float mu = 0.f;
#pragma unroll
        for (int d = 0; d < 64; ++d) mu += s_down[row][d];
        mu *= (1.f / 64.f);
        float var = 0.f;
#pragma unroll
        for (int d = 0; d < 64; ++d) {
            float t = s_down[row][d] - mu;
            var += t * t;
        }
        var *= (1.f / 64.f);
        const float rs = rsqrtf(var + EPS);
#pragma unroll
        for (int d = 0; d < 64; ++d)
            s_norm[row * 64 + d] =
                __float2bfloat16((s_down[row][d] - mu) * rs * gamma[d] + beta[d]);
    }
    __syncthreads();

    f32x8 o1[4] = {}, o2[4] = {};
#pragma unroll
    for (int k = 0; k < D_DIM; k += 32) {
        Frag a = load_frag(s_norm, 64, wave * 16, k);
#pragma unroll
        for (int t = 0; t < 4; ++t) {
            Frag u = load_frag(W1, D_DIM, 16 * t, k);
            Frag g = load_frag(W2, D_DIM, 16 * t, k);
            o1[t] = wmma_bf16(a, u, o1[t]);
            o2[t] = wmma_bf16(a, g, o2[t]);
        }
    }

    const int mb = m0 + ((lane >> 4) << 3);
#pragma unroll
    for (int t = 0; t < 4; ++t) {
        const int e   = 16 * t + (lane & 15);
        const float bb1 = b1[e];
        const float bb2 = b2[e];
#pragma unroll
        for (int r = 0; r < 8; ++r) {
            const float x  = o1[t][r] + bb1;
            const float g  = o2[t][r] + bb2;
            const float sw = x / (1.f + __expf(-x));   // x * sigmoid(x)
            Act[(size_t)(mb + r) * D_DIM + e] = __float2bfloat16(sw * g);
        }
    }
}

// ---------------------------------------------------------------------------
// K3: out[M,Q] = residual + Act[M,D] x Wup[Q,D]^T   (K=64: direct loads)
// grid = (M/128, Q/128), block = 256
// ---------------------------------------------------------------------------
__global__ void k3_up_residual(const __hip_bfloat16* __restrict__ Act,
                               const __hip_bfloat16* __restrict__ Wu,
                               const float* __restrict__ residual,
                               float* __restrict__ out) {
    const int wave = threadIdx.x >> 5;
    const int lane = threadIdx.x & 31;
    const int m0   = blockIdx.x * 128 + wave * 16;
    const int n0   = blockIdx.y * 128;

    f32x8 acc[8] = {};
#pragma unroll
    for (int k = 0; k < D_DIM; k += 32) {
        Frag a = load_frag(Act, D_DIM, m0, k);
#pragma unroll
        for (int t = 0; t < 8; ++t) {
            Frag b = load_frag(Wu, D_DIM, n0 + 16 * t, k);
            acc[t] = wmma_bf16(a, b, acc[t]);
        }
    }

    const int mb = m0 + ((lane >> 4) << 3);
#pragma unroll
    for (int t = 0; t < 8; ++t) {
        const int n = n0 + 16 * t + (lane & 15);
#pragma unroll
        for (int r = 0; r < 8; ++r) {
            const size_t idx = (size_t)(mb + r) * Q_DIM + n;
            out[idx] = residual[idx] + acc[t][r];
        }
    }
}

// ---------------------------------------------------------------------------
// Launch
// ---------------------------------------------------------------------------
extern "C" void kernel_launch(void* const* d_in, const int* in_sizes, int n_in,
                              void* d_out, int out_size, void* d_ws, size_t ws_size,
                              hipStream_t stream) {
    (void)in_sizes; (void)n_in; (void)out_size; (void)ws_size;

    const float* clamp3   = (const float*)d_in[0];   // [M, C]
    const float* residual = (const float*)d_in[1];   // [M, Q]
    const float* Wproj    = (const float*)d_in[2];   // [Q, C]
    const float* Wdown    = (const float*)d_in[3];   // [D, Q]
    const float* gamma    = (const float*)d_in[4];   // [D]
    const float* beta     = (const float*)d_in[5];   // [D]
    const float* Wl1      = (const float*)d_in[6];   // [D, D]
    const float* bl1      = (const float*)d_in[7];   // [D]
    const float* Wl2      = (const float*)d_in[8];   // [D, D]
    const float* bl2      = (const float*)d_in[9];   // [D]
    const float* Wup      = (const float*)d_in[10];  // [Q, D]
    float* out = (float*)d_out;

    char* ws = (char*)d_ws;
    size_t off = 0;
    auto carve = [&](size_t bytes) -> __hip_bfloat16* {
        void* p = ws + off;
        off = (off + bytes + 255) & ~(size_t)255;
        return (__hip_bfloat16*)p;
    };
    __hip_bfloat16* Pbf  = carve((size_t)M_TOT * Q_DIM * 2);   // 67.1 MB
    __hip_bfloat16* Actb = carve((size_t)M_TOT * D_DIM * 2);   //  4.2 MB
    __hip_bfloat16* Wpb  = carve((size_t)Q_DIM * C_DIM * 2);
    __hip_bfloat16* Wdb  = carve((size_t)D_DIM * Q_DIM * 2);
    __hip_bfloat16* W1b  = carve((size_t)D_DIM * D_DIM * 2);
    __hip_bfloat16* W2b  = carve((size_t)D_DIM * D_DIM * 2);
    __hip_bfloat16* Wub  = carve((size_t)Q_DIM * D_DIM * 2);

    auto cvt = [&](const float* src, __hip_bfloat16* dst, int n) {
        cvt_f32_bf16<<<(n + 255) / 256, 256, 0, stream>>>(src, dst, n);
    };
    cvt(Wproj, Wpb, Q_DIM * C_DIM);
    cvt(Wdown, Wdb, D_DIM * Q_DIM);
    cvt(Wl1,   W1b, D_DIM * D_DIM);
    cvt(Wl2,   W2b, D_DIM * D_DIM);
    cvt(Wup,   Wub, Q_DIM * D_DIM);

    k1_proj<<<dim3(M_TOT / 128, Q_DIM / 128), 256, 0, stream>>>(clamp3, Wpb, Pbf);

    k2_down_ln_swiglu<<<dim3(M_TOT / 128), 256, 0, stream>>>(
        Pbf, Wdb, gamma, beta, W1b, bl1, W2b, bl2, Actb);

    k3_up_residual<<<dim3(M_TOT / 128, Q_DIM / 128), 256, 0, stream>>>(
        Actb, Wub, residual, out);
}
